// BatchTopKSAE_24885040513333
// MI455X (gfx1250) — matrix operand order repaired and checked by reading, not compiled, not run on hardware
//
#include <hip/hip_runtime.h>

// ---------------------------------------------------------------------------
// BatchTopK SAE forward for MI455X (gfx1250, wave32, WMMA).
//   encode  : latent = x @ W_enc + b_enc          (split-bf16 WMMA, f32 accum)
//   topk    : threshold via 2-level radix histogram on score keys
//   mask    : sparse = latent * (score >= thr)
//   decode  : recon  = sparse @ W_dec + b_dec     (split-bf16 WMMA, f32 accum)
// ---------------------------------------------------------------------------

#define SAE_BATCH   4096
#define SAE_DMODEL  2048
#define SAE_DICT    32768
#define SAE_NTOT    ((long long)SAE_BATCH * SAE_DICT)

typedef __attribute__((ext_vector_type(16))) __bf16 v16bf;
typedef __attribute__((ext_vector_type(8)))  float  v8f;

union FragBF { v16bf v; uint4 q[2]; };

__device__ __forceinline__ unsigned short f2bfu(float f) {
  // round-to-nearest-even f32 -> bf16 bit pattern
  unsigned u = __float_as_uint(f);
  unsigned r = u + 0x7FFFu + ((u >> 16) & 1u);
  return (unsigned short)(r >> 16);
}
__device__ __forceinline__ float bfu2f(unsigned short h) {
  return __uint_as_float((unsigned)h << 16);
}
__device__ __forceinline__ unsigned fkey(float s) {
  // monotone float -> uint mapping (descending score == descending key)
  unsigned u = __float_as_uint(s);
  return (u & 0x80000000u) ? ~u : (u | 0x80000000u);
}
__device__ __forceinline__ unsigned pack2(unsigned short lo, unsigned short hi) {
  return (unsigned)lo | ((unsigned)hi << 16);
}

// ---------------------------------------------------------------------------
// Split-bf16 GEMM: C[M,N] = A[M,K] @ B[K,N] + bias[N]
// Block tile 64(M) x 128(N), K-step 32, 256 threads = 8 waves (2x4),
// each wave owns a 32x32 patch = 2x2 WMMA 16x16 tiles.
// Each f32 is decomposed hi+lo bf16; product uses 3 WMMAs (drops lo*lo).
// Double-buffered LDS: one barrier per K-step; packed DS stores (b128/b32).
// ---------------------------------------------------------------------------
__global__ __launch_bounds__(256)
void sae_gemm_bias(const float* __restrict__ A, const float* __restrict__ B,
                   const float* __restrict__ bias, float* __restrict__ C,
                   int M, int N, int K)
{
  __shared__ __align__(16) unsigned short AsmH[2][64][40];
  __shared__ __align__(16) unsigned short AsmL[2][64][40];
  __shared__ __align__(16) unsigned short BsmH[2][128][40];  // transposed: [col][k]
  __shared__ __align__(16) unsigned short BsmL[2][128][40];

  const int t    = threadIdx.x;
  const int lane = t & 31;
  const int wid  = t >> 5;
  const int wm   = wid >> 2;            // 0..1  (M direction)
  const int wn   = wid & 3;             // 0..3  (N direction)
  const int rowbase = blockIdx.x * 64;
  const int colbase = blockIdx.y * 128;

  // cooperative global->LDS load indices
  const int arow = t >> 2;              // 0..63, 8 floats each (one k-run)
  const int acol = (t & 3) * 8;
  const int bkp  = (t >> 4) * 2;        // k-row pair: 0,2,..,30
  const int bcol = (t & 15) * 8;        // 8 columns each

  const int kh  = lane >> 4;            // lane half
  const int l15 = lane & 15;

  v8f acc[2][2] = {};

  int p = 0;
  for (int k0 = 0; k0 < K; k0 += 32, p ^= 1) {
    const float4* ap = (const float4*)(A + (size_t)(rowbase + arow) * K + (k0 + acol));
    float4 av0 = ap[0], av1 = ap[1];
    const float4* bp0 = (const float4*)(B + (size_t)(k0 + bkp) * N + (colbase + bcol));
    const float4* bp1 = (const float4*)((const float*)bp0 + N);
    float4 b0a = bp0[0], b0b = bp0[1];
    float4 b1a = bp1[0], b1b = bp1[1];
    if (k0 + 32 < K) {
      __builtin_prefetch(A + (size_t)(rowbase + arow) * K + (k0 + 32 + acol), 0, 1);
      __builtin_prefetch(B + (size_t)(k0 + 32 + bkp) * N + (colbase + bcol), 0, 1);
    }
    {
      // --- A tile: 8 contiguous k per thread -> one b128 store each (hi, lo)
      float av[8] = {av0.x, av0.y, av0.z, av0.w, av1.x, av1.y, av1.z, av1.w};
      unsigned short h[8], l[8];
#pragma unroll
      for (int j = 0; j < 8; ++j) {
        h[j] = f2bfu(av[j]);
        l[j] = f2bfu(av[j] - bfu2f(h[j]));
      }
      uint4 uh, ul;
      uh.x = pack2(h[0], h[1]); uh.y = pack2(h[2], h[3]);
      uh.z = pack2(h[4], h[5]); uh.w = pack2(h[6], h[7]);
      ul.x = pack2(l[0], l[1]); ul.y = pack2(l[2], l[3]);
      ul.z = pack2(l[4], l[5]); ul.w = pack2(l[6], l[7]);
      *(uint4*)&AsmH[p][arow][acol] = uh;
      *(uint4*)&AsmL[p][arow][acol] = ul;

      // --- B tile: 2 k-rows x 8 cols per thread -> packed b32 stores (transposed)
      float r0[8] = {b0a.x, b0a.y, b0a.z, b0a.w, b0b.x, b0b.y, b0b.z, b0b.w};
      float r1[8] = {b1a.x, b1a.y, b1a.z, b1a.w, b1b.x, b1b.y, b1b.z, b1b.w};
#pragma unroll
      for (int j = 0; j < 8; ++j) {
        unsigned short h0 = f2bfu(r0[j]);
        unsigned short h1 = f2bfu(r1[j]);
        *(unsigned*)&BsmH[p][bcol + j][bkp] = pack2(h0, h1);
        unsigned short l0 = f2bfu(r0[j] - bfu2f(h0));
        unsigned short l1 = f2bfu(r1[j] - bfu2f(h1));
        *(unsigned*)&BsmL[p][bcol + j][bkp] = pack2(l0, l1);
      }
    }
    __syncthreads();

    // Assemble fragments from LDS per the 16-bit WMMA VGPR layouts.
    // A 16x32: lane(l15,kh) holds K = {kh*8..+7} U {16+kh*8..+7} for row l15.
    // B 32x16: lane(l15,kh) holds K = {kh*8..+7} U {16+kh*8..+7} for column l15.
    FragBF faH[2], faL[2], fbH[2], fbL[2];
#pragma unroll
    for (int tm = 0; tm < 2; ++tm) {
      int r0 = wm * 32 + tm * 16 + l15;
      faH[tm].q[0] = *(const uint4*)&AsmH[p][r0][kh * 8];
      faH[tm].q[1] = *(const uint4*)&AsmH[p][r0][16 + kh * 8];
      faL[tm].q[0] = *(const uint4*)&AsmL[p][r0][kh * 8];
      faL[tm].q[1] = *(const uint4*)&AsmL[p][r0][16 + kh * 8];
    }
#pragma unroll
    for (int tn = 0; tn < 2; ++tn) {
      int c0 = wn * 32 + tn * 16 + l15;
      fbH[tn].q[0] = *(const uint4*)&BsmH[p][c0][kh * 8];
      fbH[tn].q[1] = *(const uint4*)&BsmH[p][c0][16 + kh * 8];
      fbL[tn].q[0] = *(const uint4*)&BsmL[p][c0][kh * 8];
      fbL[tn].q[1] = *(const uint4*)&BsmL[p][c0][16 + kh * 8];
    }
#pragma unroll
    for (int tm = 0; tm < 2; ++tm)
#pragma unroll
      for (int tn = 0; tn < 2; ++tn) {
        v8f c = acc[tm][tn];
        c = __builtin_amdgcn_wmma_f32_16x16x32_bf16(false, faH[tm].v, false, fbH[tn].v,
                                                    (short)0, c, false, false);
        c = __builtin_amdgcn_wmma_f32_16x16x32_bf16(false, faL[tm].v, false, fbH[tn].v,
                                                    (short)0, c, false, false);
        c = __builtin_amdgcn_wmma_f32_16x16x32_bf16(false, faH[tm].v, false, fbL[tn].v,
                                                    (short)0, c, false, false);
        acc[tm][tn] = c;
      }
  }

  // Epilogue: C/D layout -> VGPR r holds M = kh*8 + r, N = l15.
#pragma unroll
  for (int tm = 0; tm < 2; ++tm) {
    int grow = rowbase + wm * 32 + tm * 16 + kh * 8;
#pragma unroll
    for (int tn = 0; tn < 2; ++tn) {
      int gcol = colbase + wn * 32 + tn * 16 + l15;
      float bb = bias[gcol];
#pragma unroll
      for (int r = 0; r < 8; ++r)
        C[(size_t)(grow + r) * N + gcol] = acc[tm][tn][r] + bb;
    }
  }
}

// ---------------------------------------------------------------------------
// decoder_norms[f] = || W_dec[f, :] ||_2   (one wave32 per feature row)
// ---------------------------------------------------------------------------
__global__ __launch_bounds__(256)
void sae_norms(const float* __restrict__ Wdec, float* __restrict__ norms)
{
  int gw   = (int)((blockIdx.x * blockDim.x + threadIdx.x) >> 5);
  int lane = threadIdx.x & 31;
  if (gw >= SAE_DICT) return;
  const float4* row = (const float4*)(Wdec + (size_t)gw * SAE_DMODEL);
  float s = 0.f;
  for (int j = lane; j < SAE_DMODEL / 4; j += 32) {
    float4 v = row[j];
    s += v.x * v.x + v.y * v.y + v.z * v.z + v.w * v.w;
  }
  for (int o = 16; o > 0; o >>= 1) s += __shfl_xor(s, o, 32);
  if (lane == 0) norms[gw] = sqrtf(s);
}

// ---------------------------------------------------------------------------
// histograms over score keys.  pass 0: top-12-bit bins (LDS privatized).
// pass 1: low-20-bit bins restricted to the crossing coarse bin (global).
// ---------------------------------------------------------------------------
__global__ __launch_bounds__(256)
void sae_hist(const float* __restrict__ latent, const float* __restrict__ norms,
              unsigned* __restrict__ h1, unsigned* __restrict__ h2,
              const unsigned* __restrict__ ctrl, int pass, long long n4)
{
  __shared__ unsigned lh[4096];
  if (pass == 0) {
    for (int i = threadIdx.x; i < 4096; i += 256) lh[i] = 0u;
    __syncthreads();
  }
  unsigned binT = pass ? ctrl[0] : 0u;
  const float4* lat4 = (const float4*)latent;
  long long stride = (long long)gridDim.x * blockDim.x;
  for (long long i = (long long)blockIdx.x * blockDim.x + threadIdx.x; i < n4; i += stride) {
    float4 v = lat4[i];
    int f0 = (int)((i * 4) & (SAE_DICT - 1));
    float s[4] = {v.x * norms[f0], v.y * norms[f0 + 1],
                  v.z * norms[f0 + 2], v.w * norms[f0 + 3]};
#pragma unroll
    for (int j = 0; j < 4; ++j) {
      unsigned u = fkey(s[j]);
      if (pass == 0)              atomicAdd(&lh[u >> 20], 1u);
      else if ((u >> 20) == binT) atomicAdd(&h2[u & 0xFFFFFu], 1u);
    }
  }
  if (pass == 0) {
    __syncthreads();
    for (int i = threadIdx.x; i < 4096; i += 256) {
      unsigned c = lh[i];
      if (c) atomicAdd(&h1[i], c);
    }
  }
}

// ---------------------------------------------------------------------------
// single-block descending scan: find bin where cumulative count crosses target.
// pass 0: target = k*BATCH over h1 (4096 bins) -> ctrl[0]=binT, ctrl[1]=remaining
// pass 1: target = ctrl[1]  over h2 (2^20 bins) -> ctrl[2]=threshold key
// ---------------------------------------------------------------------------
__global__ __launch_bounds__(256)
void sae_scan(const unsigned* __restrict__ hist, int nbins,
              const int* __restrict__ kptr, unsigned* __restrict__ ctrl, int pass)
{
  __shared__ unsigned partial[256];
  int t = threadIdx.x;
  int per = nbins >> 8;
  unsigned s = 0;
  for (int j = 0; j < per; ++j) s += hist[t * per + j];
  partial[t] = s;
  __syncthreads();
  if (t == 0) {
    unsigned long long target =
        pass ? (unsigned long long)ctrl[1]
             : (unsigned long long)kptr[0] * (unsigned long long)SAE_BATCH;
    if (target < 1) target = 1;
    unsigned long long cum = 0;
    int seg = 0;
    for (int sg = 255; sg >= 0; --sg) {
      unsigned p = partial[sg];
      if (cum + p >= target || sg == 0) { seg = sg; break; }
      cum += p;
    }
    int binT = seg * per;
    unsigned long long rem = 1;
    for (int b = seg * per + per - 1; b >= seg * per; --b) {
      unsigned c = hist[b];
      if (cum + c >= target || b == seg * per) {
        binT = b;
        rem  = (target > cum) ? (target - cum) : 1ull;
        break;
      }
      cum += c;
    }
    if (pass == 0) { ctrl[0] = (unsigned)binT; ctrl[1] = (unsigned)rem; }
    else           { ctrl[2] = (ctrl[0] << 20) | (unsigned)binT; }
  }
}

// ---------------------------------------------------------------------------
// sparse = latent where key(score) >= threshold else 0   (float4 streaming)
// ---------------------------------------------------------------------------
__global__ __launch_bounds__(256)
void sae_mask(const float* __restrict__ latent, const float* __restrict__ norms,
              const unsigned* __restrict__ ctrl, float* __restrict__ sparse,
              long long n4)
{
  unsigned thr = ctrl[2];
  const float4* lat4 = (const float4*)latent;
  float4* sp4 = (float4*)sparse;
  long long stride = (long long)gridDim.x * blockDim.x;
  for (long long i = (long long)blockIdx.x * blockDim.x + threadIdx.x; i < n4; i += stride) {
    float4 v = lat4[i];
    int f0 = (int)((i * 4) & (SAE_DICT - 1));
    float4 o;
    o.x = (fkey(v.x * norms[f0    ]) >= thr) ? v.x : 0.0f;
    o.y = (fkey(v.y * norms[f0 + 1]) >= thr) ? v.y : 0.0f;
    o.z = (fkey(v.z * norms[f0 + 2]) >= thr) ? v.z : 0.0f;
    o.w = (fkey(v.w * norms[f0 + 3]) >= thr) ? v.w : 0.0f;
    sp4[i] = o;
  }
}

__global__ __launch_bounds__(256)
void sae_zero(unsigned* __restrict__ p, long long n)
{
  long long stride = (long long)gridDim.x * blockDim.x;
  for (long long i = (long long)blockIdx.x * blockDim.x + threadIdx.x; i < n; i += stride)
    p[i] = 0u;
}

// ---------------------------------------------------------------------------
extern "C" void kernel_launch(void* const* d_in, const int* in_sizes, int n_in,
                              void* d_out, int out_size, void* d_ws, size_t ws_size,
                              hipStream_t stream) {
  (void)in_sizes; (void)n_in; (void)out_size; (void)ws_size;

  const float* x     = (const float*)d_in[0];   // [4096, 2048]
  const float* W_enc = (const float*)d_in[1];   // [2048, 32768]
  const float* b_enc = (const float*)d_in[2];   // [32768]
  const float* W_dec = (const float*)d_in[3];   // [32768, 2048]
  const float* b_dec = (const float*)d_in[4];   // [2048]
  const int*   kptr  = (const int*)d_in[5];     // scalar k (device)

  // d_out = reconstruction | sparse | latent (return order, flat)
  float* recon  = (float*)d_out;
  float* sparse = recon  + (size_t)SAE_BATCH * SAE_DMODEL;
  float* latent = sparse + (size_t)SAE_BATCH * SAE_DICT;

  // workspace layout (4-byte units)
  float*    norms = (float*)d_ws;                       // 32768
  unsigned* h1    = (unsigned*)d_ws + SAE_DICT;         // 4096
  unsigned* h2    = h1 + 4096;                          // 1048576
  unsigned* ctrl  = h2 + 1048576;                       // 4

  const long long nzero = 4096 + 1048576 + 4;

  sae_zero <<<1024, 256, 0, stream>>>(h1, nzero);
  sae_norms<<<SAE_DICT / 8, 256, 0, stream>>>(W_dec, norms);

  // encode: latent = x @ W_enc + b_enc   (M=4096, N=32768, K=2048)
  sae_gemm_bias<<<dim3(SAE_BATCH / 64, SAE_DICT / 128), 256, 0, stream>>>(
      x, W_enc, b_enc, latent, SAE_BATCH, SAE_DICT, SAE_DMODEL);

  // batch top-k threshold (exact 32-bit key via 12+20-bit radix passes)
  sae_hist<<<4096, 256, 0, stream>>>(latent, norms, h1, h2, ctrl, 0, SAE_NTOT / 4);
  sae_scan<<<1, 256, 0, stream>>>(h1, 4096, kptr, ctrl, 0);
  sae_hist<<<4096, 256, 0, stream>>>(latent, norms, h1, h2, ctrl, 1, SAE_NTOT / 4);
  sae_scan<<<1, 256, 0, stream>>>(h2, 1 << 20, kptr, ctrl, 1);

  sae_mask<<<4096, 256, 0, stream>>>(latent, norms, ctrl, sparse, SAE_NTOT / 4);

  // decode: recon = sparse @ W_dec + b_dec  (M=4096, N=2048, K=32768)
  sae_gemm_bias<<<dim3(SAE_BATCH / 64, SAE_DMODEL / 128), 256, 0, stream>>>(
      sparse, W_dec, b_dec, recon, SAE_BATCH, SAE_DMODEL, SAE_DICT);
}
